// ResidualLSTMCell_68092411511372
// MI455X (gfx1250) — matrix operand order repaired and checked by reading, not compiled, not run on hardware
//
#include <hip/hip_runtime.h>
#include <math.h>

// ---------------------------------------------------------------------------
// Residual peephole-LSTM with projection, MI455X (gfx1250, wave32, WMMA).
// All matmuls use v_wmma_f32_16x16x32_bf16 (bf16 in, f32 accumulate).
// ---------------------------------------------------------------------------

typedef __attribute__((ext_vector_type(16))) __bf16 v16bf;
typedef __attribute__((ext_vector_type(8)))  __bf16 v8bf;
typedef __attribute__((ext_vector_type(8)))  float  v8f;

#define B_  64
#define T_  512
#define D_  1024
#define U_  1024
#define G4_ 4096   // 4*U

// float -> bf16 bits, round-to-nearest-even
__device__ __forceinline__ unsigned short f2bf(float f) {
  union { float f; unsigned int u; } v; v.f = f;
  unsigned int u = v.u;
  u += 0x7FFFu + ((u >> 16) & 1u);
  return (unsigned short)(u >> 16);
}

__device__ __forceinline__ float hsig(float x) {
  return fminf(fmaxf(0.2f * x + 0.5f, 0.0f), 1.0f);
}

union afrag_u { v16bf v; v8bf h[2]; };

// A-fragment (16x32 bf16, one row per lane pair-group):
// lanes 0-15 : K = k0+0..7  and k0+16..23
// lanes16-31 : K = k0+8..15 and k0+24..31
// caller passes p = row + k0 + (lane>=16 ? 8 : 0); chunks at p and p+16.
__device__ __forceinline__ v16bf load_a16(const unsigned short* p) {
  afrag_u u;
  u.h[0] = *(const v8bf*)(p);
  u.h[1] = *(const v8bf*)(p + 16);
  return u.v;
}

// B-fragment (32x16 bf16) from pre-transposed weights (N x K row-major):
// lanes 0-15: N=lane, K=k0+0..7 and k0+16..23 ; lanes16-31: N=lane-16, K=k0+8..15 and k0+24..31
// -> two 16B loads per lane at p and p+16.
__device__ __forceinline__ v16bf load_b16(const unsigned short* p) {
  afrag_u u;
  u.h[0] = *(const v8bf*)(p);
  u.h[1] = *(const v8bf*)(p + 16);
  return u.v;
}

// ---------------------------------------------------------------------------
// Phase 0a: transpose f32 (K x N) -> bf16 (N x K), LDS-tiled for coalescing.
// ---------------------------------------------------------------------------
__global__ void transpose_f32_to_bf16(const float* __restrict__ src,
                                      unsigned short* __restrict__ dst,
                                      int K, int N) {
  __shared__ float tile[32][33];
  int n0 = blockIdx.x * 32;
  int k0 = blockIdx.y * 32;
  int tx = threadIdx.x;      // 0..31
  int ty = threadIdx.y;      // 0..7
  for (int i = ty; i < 32; i += 8)
    tile[i][tx] = src[(size_t)(k0 + i) * N + (n0 + tx)];
  __syncthreads();
  for (int i = ty; i < 32; i += 8)
    dst[(size_t)(n0 + i) * K + (k0 + tx)] = f2bf(tile[tx][i]);
}

// ---------------------------------------------------------------------------
// Phase 0b: initial state h0/c0 -> bf16 h, f32+bf16 c
// ---------------------------------------------------------------------------
__global__ void init_state(const float* __restrict__ h0,
                           const float* __restrict__ c0,
                           unsigned short* __restrict__ h_bf,
                           unsigned short* __restrict__ c_bf,
                           float* __restrict__ c_f) {
  int i = blockIdx.x * blockDim.x + threadIdx.x;
  if (i < B_ * U_) {
    h_bf[i] = f2bf(h0[i]);
    float c = c0[i];
    c_bf[i] = f2bf(c);
    c_f[i]  = c;
  }
}

// ---------------------------------------------------------------------------
// Phase 1: ZX = X @ W + bias   (32768 x 4096, K=1024), f32 out.
// One wave = one 16-row M-tile x 4 N-tiles (A fragment reused 4x).
// ---------------------------------------------------------------------------
__global__ __launch_bounds__(256) void gemm_zx(
    const float* __restrict__ X,            // (B*T, D) row-major
    const unsigned short* __restrict__ WkT, // (4U, D) bf16
    const float* __restrict__ bias,         // (4U)
    float* __restrict__ ZX) {               // (B*T, 4U)
  int wave = blockIdx.x * (blockDim.x >> 5) + (threadIdx.x >> 5);
  int lane = threadIdx.x & 31;
  int mt   = wave >> 6;        // 0..2047
  int ng   = wave & 63;        // 0..63 -> n-tiles 4*ng..4*ng+3
  int half = lane >> 4;
  int lm   = lane & 15;

  const float* arow = X + (size_t)(mt * 16 + lm) * D_;
  v8f acc0 = {}, acc1 = {}, acc2 = {}, acc3 = {};

  for (int k0 = 0; k0 < D_; k0 += 32) {
    // A fragment: convert f32 -> bf16 on the fly
    v16bf a;
    const float* pa = arow + k0 + half * 8;
    afrag_u ua;
    unsigned short tmp[16];
#pragma unroll
    for (int j = 0; j < 8; ++j) tmp[j]     = f2bf(pa[j]);
#pragma unroll
    for (int j = 0; j < 8; ++j) tmp[8 + j] = f2bf(pa[16 + j]);
    ua.h[0] = *(const v8bf*)(&tmp[0]);
    ua.h[1] = *(const v8bf*)(&tmp[8]);
    a = ua.v;

    int ofsB = k0 + half * 8;
    const unsigned short* pb0 = WkT + (size_t)((ng * 4 + 0) * 16 + lm) * D_ + ofsB;
    const unsigned short* pb1 = WkT + (size_t)((ng * 4 + 1) * 16 + lm) * D_ + ofsB;
    const unsigned short* pb2 = WkT + (size_t)((ng * 4 + 2) * 16 + lm) * D_ + ofsB;
    const unsigned short* pb3 = WkT + (size_t)((ng * 4 + 3) * 16 + lm) * D_ + ofsB;
    acc0 = __builtin_amdgcn_wmma_f32_16x16x32_bf16(false, a, false, load_b16(pb0), (short)0, acc0, false, false);
    acc1 = __builtin_amdgcn_wmma_f32_16x16x32_bf16(false, a, false, load_b16(pb1), (short)0, acc1, false, false);
    acc2 = __builtin_amdgcn_wmma_f32_16x16x32_bf16(false, a, false, load_b16(pb2), (short)0, acc2, false, false);
    acc3 = __builtin_amdgcn_wmma_f32_16x16x32_bf16(false, a, false, load_b16(pb3), (short)0, acc3, false, false);
  }

  int row0 = mt * 16 + half * 8;
  v8f* accs[4] = { &acc0, &acc1, &acc2, &acc3 };
#pragma unroll
  for (int nt = 0; nt < 4; ++nt) {
    int n = (ng * 4 + nt) * 16 + lm;
    float bv = bias[n];
    float* op = ZX + (size_t)row0 * G4_ + n;
#pragma unroll
    for (int j = 0; j < 8; ++j)
      op[(size_t)j * G4_] = (*accs[nt])[j] + bv;
  }
}

// ---------------------------------------------------------------------------
// Step A: per wave (mt 0..3, nt 0..63): compute the 4 gate z-tiles of h@R and
// 2 peephole tiles of c@P, add ZX, do i/f gates and the c update.
// ---------------------------------------------------------------------------
__global__ __launch_bounds__(256) void step_a(
    int t,
    const float* __restrict__ ZX,
    const unsigned short* __restrict__ RT,   // (4U, U) bf16
    const unsigned short* __restrict__ PT,   // (3U, U) bf16
    const unsigned short* __restrict__ h_bf, // (B, U)
    const unsigned short* __restrict__ c_bf_rd,
    const float* __restrict__ c_f_rd,
    unsigned short* __restrict__ c_bf_wr,
    float* __restrict__ c_f_wr,
    unsigned short* __restrict__ tanhc_bf,
    float* __restrict__ zo_buf) {
  int wave = blockIdx.x * (blockDim.x >> 5) + (threadIdx.x >> 5);
  int lane = threadIdx.x & 31;
  int mt   = wave >> 6;        // 0..3   (batch tile)
  int nt   = wave & 63;        // 0..63  (gate-column tile)
  int half = lane >> 4;
  int lm   = lane & 15;
  int col  = nt * 16 + lm;     // 0..1023

  const unsigned short* hrow = h_bf    + (size_t)(mt * 16 + lm) * U_;
  const unsigned short* crow = c_bf_rd + (size_t)(mt * 16 + lm) * U_;
  const unsigned short* rti = RT + (size_t)(col)          * U_;
  const unsigned short* rtf = RT + (size_t)(col +     U_) * U_;
  const unsigned short* rtc = RT + (size_t)(col + 2 * U_) * U_;
  const unsigned short* rto = RT + (size_t)(col + 3 * U_) * U_;
  const unsigned short* pti = PT + (size_t)(col)          * U_;
  const unsigned short* ptf = PT + (size_t)(col +     U_) * U_;

  v8f zi = {}, zf = {}, zc = {}, zo = {}, cpi = {}, cpf = {};
  for (int k0 = 0; k0 < U_; k0 += 32) {
    int ofsA = k0 + half * 8;
    int ofsB = k0 + half * 8;
    v16bf ah = load_a16(hrow + ofsA);
    v16bf ac = load_a16(crow + ofsA);
    zi  = __builtin_amdgcn_wmma_f32_16x16x32_bf16(false, ah, false, load_b16(rti + ofsB), (short)0, zi,  false, false);
    zf  = __builtin_amdgcn_wmma_f32_16x16x32_bf16(false, ah, false, load_b16(rtf + ofsB), (short)0, zf,  false, false);
    zc  = __builtin_amdgcn_wmma_f32_16x16x32_bf16(false, ah, false, load_b16(rtc + ofsB), (short)0, zc,  false, false);
    zo  = __builtin_amdgcn_wmma_f32_16x16x32_bf16(false, ah, false, load_b16(rto + ofsB), (short)0, zo,  false, false);
    cpi = __builtin_amdgcn_wmma_f32_16x16x32_bf16(false, ac, false, load_b16(pti + ofsB), (short)0, cpi, false, false);
    cpf = __builtin_amdgcn_wmma_f32_16x16x32_bf16(false, ac, false, load_b16(ptf + ofsB), (short)0, cpf, false, false);
  }

  int b0 = mt * 16 + half * 8;
#pragma unroll
  for (int j = 0; j < 8; ++j) {
    int brow = b0 + j;
    size_t zxbase = ((size_t)brow * T_ + t) * (size_t)G4_;
    float vi = zi[j] + ZX[zxbase + col]          + cpi[j];
    float vf = zf[j] + ZX[zxbase + col +     U_] + cpf[j];
    float vc = zc[j] + ZX[zxbase + col + 2 * U_];
    float vo = zo[j] + ZX[zxbase + col + 3 * U_];
    float ig = hsig(vi);
    float fg = hsig(vf);
    size_t idx = (size_t)brow * U_ + col;
    float cn = fg * c_f_rd[idx] + ig * tanhf(vc);
    c_f_wr[idx]   = cn;
    c_bf_wr[idx]  = f2bf(cn);
    tanhc_bf[idx] = f2bf(tanhf(cn));
    zo_buf[idx]   = vo;
  }
}

// ---------------------------------------------------------------------------
// Step B: per wave: c_new@Po and tanh(c_new)@Wproj tiles, o gate, residual,
// write out[:, t, :] and bf16 h for next step.
// ---------------------------------------------------------------------------
__global__ __launch_bounds__(256) void step_b(
    int t,
    const float* __restrict__ X,               // (B, T, D)
    const unsigned short* __restrict__ PT,     // (3U, U)
    const unsigned short* __restrict__ WpT,    // (U, U)
    const unsigned short* __restrict__ cnew_bf,
    const unsigned short* __restrict__ tanhc_bf,
    const float* __restrict__ zo_buf,
    unsigned short* __restrict__ h_bf,
    float* __restrict__ out) {
  int wave = blockIdx.x * (blockDim.x >> 5) + (threadIdx.x >> 5);
  int lane = threadIdx.x & 31;
  int mt   = wave >> 6;
  int nt   = wave & 63;
  int half = lane >> 4;
  int lm   = lane & 15;
  int col  = nt * 16 + lm;

  const unsigned short* arc = cnew_bf  + (size_t)(mt * 16 + lm) * U_;
  const unsigned short* art = tanhc_bf + (size_t)(mt * 16 + lm) * U_;
  const unsigned short* pto = PT  + (size_t)(col + 2 * U_) * U_;
  const unsigned short* wpt = WpT + (size_t)(col)          * U_;

  v8f po = {}, pr = {};
  for (int k0 = 0; k0 < U_; k0 += 32) {
    int ofsA = k0 + half * 8;
    int ofsB = k0 + half * 8;
    v16bf ac = load_a16(arc + ofsA);
    v16bf at = load_a16(art + ofsA);
    po = __builtin_amdgcn_wmma_f32_16x16x32_bf16(false, ac, false, load_b16(pto + ofsB), (short)0, po, false, false);
    pr = __builtin_amdgcn_wmma_f32_16x16x32_bf16(false, at, false, load_b16(wpt + ofsB), (short)0, pr, false, false);
  }

  int b0 = mt * 16 + half * 8;
#pragma unroll
  for (int j = 0; j < 8; ++j) {
    int brow = b0 + j;
    size_t idx = (size_t)brow * U_ + col;
    float o  = hsig(zo_buf[idx] + po[j]);
    float xr = X[((size_t)brow * T_ + t) * D_ + col];
    float h  = o * (pr[j] + xr);
    out[((size_t)brow * T_ + t) * U_ + col] = h;
    h_bf[idx] = f2bf(h);
  }
}

// ---------------------------------------------------------------------------
// Host launcher
// ---------------------------------------------------------------------------
extern "C" void kernel_launch(void* const* d_in, const int* in_sizes, int n_in,
                              void* d_out, int out_size, void* d_ws, size_t ws_size,
                              hipStream_t stream) {
  (void)in_sizes; (void)n_in; (void)out_size; (void)ws_size;
  const float* x    = (const float*)d_in[0];
  const float* h0   = (const float*)d_in[1];
  const float* c0   = (const float*)d_in[2];
  const float* Wk   = (const float*)d_in[3];
  const float* Rk   = (const float*)d_in[4];
  const float* Pk   = (const float*)d_in[5];
  const float* Wp   = (const float*)d_in[6];
  const float* bias = (const float*)d_in[7];
  float* out = (float*)d_out;

  char* ws = (char*)d_ws;
  size_t off = 0;
  auto take = [&](size_t bytes) -> char* {
    char* p = ws + off;
    off = (off + bytes + 255) & ~(size_t)255;
    return p;
  };
  unsigned short* WkT   = (unsigned short*)take((size_t)G4_ * D_ * 2);
  unsigned short* RT    = (unsigned short*)take((size_t)G4_ * U_ * 2);
  unsigned short* PT    = (unsigned short*)take((size_t)3 * U_ * U_ * 2);
  unsigned short* WpT   = (unsigned short*)take((size_t)U_ * U_ * 2);
  unsigned short* h_bf  = (unsigned short*)take((size_t)B_ * U_ * 2);
  unsigned short* c_bf0 = (unsigned short*)take((size_t)B_ * U_ * 2);
  unsigned short* c_bf1 = (unsigned short*)take((size_t)B_ * U_ * 2);
  float*          c_f0  = (float*)take((size_t)B_ * U_ * 4);
  float*          c_f1  = (float*)take((size_t)B_ * U_ * 4);
  unsigned short* tanhc = (unsigned short*)take((size_t)B_ * U_ * 2);
  float*          zo    = (float*)take((size_t)B_ * U_ * 4);
  float*          ZX    = (float*)take((size_t)B_ * T_ * G4_ * 4);

  unsigned short* c_bf[2] = { c_bf0, c_bf1 };
  float*          c_f[2]  = { c_f0,  c_f1  };

  dim3 tb(32, 8);
  transpose_f32_to_bf16<<<dim3(G4_ / 32, D_ / 32), tb, 0, stream>>>(Wk, WkT, D_, G4_);
  transpose_f32_to_bf16<<<dim3(G4_ / 32, U_ / 32), tb, 0, stream>>>(Rk, RT, U_, G4_);
  transpose_f32_to_bf16<<<dim3(3 * U_ / 32, U_ / 32), tb, 0, stream>>>(Pk, PT, U_, 3 * U_);
  transpose_f32_to_bf16<<<dim3(U_ / 32, U_ / 32), tb, 0, stream>>>(Wp, WpT, U_, U_);

  init_state<<<(B_ * U_ + 255) / 256, 256, 0, stream>>>(h0, c0, h_bf, c_bf[0], c_f[0]);

  // Big time-parallel GEMM: 2048 M-tiles x 64 N-groups = 131072 waves
  gemm_zx<<<131072 / 8, 256, 0, stream>>>(x, WkT, bias, ZX);

  for (int t = 0; t < T_; ++t) {
    int rd = t & 1, wr = rd ^ 1;
    step_a<<<32, 256, 0, stream>>>(t, ZX, RT, PT, h_bf,
                                   c_bf[rd], c_f[rd], c_bf[wr], c_f[wr],
                                   tanhc, zo);
    step_b<<<32, 256, 0, stream>>>(t, x, PT, WpT,
                                   c_bf[wr], tanhc, zo, h_bf, out);
  }
}
